// ETALayer_11974368821328
// MI455X (gfx1250) — matrix operand order repaired and checked by reading, not compiled, not run on hardware
//
#include <hip/hip_runtime.h>
#include <hip/hip_bf16.h>
#include <math.h>

#define B_    1024
#define LS_   128
#define LL_   2048
#define E_    16
#define DIT_  48      // 3*E
#define TOPK_ 48
#define NH_   8
#define KD_   6
#define DIN_  224     // 48 + 80 + 48 + 48

typedef __attribute__((ext_vector_type(16))) __bf16 v16bf;
typedef __attribute__((ext_vector_type(8)))  float  v8f;

// ---------------------------------------------------------------- utilities

__device__ inline unsigned wave_ballot32(int pred) {
#if defined(__has_builtin) && __has_builtin(__builtin_amdgcn_ballot_w32)
  return __builtin_amdgcn_ballot_w32(pred != 0);
#else
  return (unsigned)__ballot(pred);
#endif
}

// A fragment: 16x32 bf16 tile from LDS rows (row-major, stride ldr, PRE-PADDED
// with zeros so every access is in-bounds -> branch-free ds_load_b64 + cvt_pk).
// CDNA5 layout: lane L -> M = L%16 ; VGPR v -> K = kbase + (v/4)*16 + (L/16)*8 + (v%4)*2 (+0,+1)
__device__ inline v16bf load_a_frag(const float* rows, int ldr, int kbase, int lane) {
  int m = lane & 15, half = lane >> 4;
  const float* base = rows + m * ldr + kbase + half * 8;
  v16bf a;
#pragma unroll
  for (int v = 0; v < 8; ++v) {
    int k = (v >> 2) * 16 + (v & 3) * 2;           // even, 8B aligned (ldr even)
    float2 x = *(const float2*)(base + k);
    a[2 * v]     = (__bf16)x.x;
    a[2 * v + 1] = (__bf16)x.y;
  }
  return a;
}

// B fragment: prepacked lane-major (lane*16 + idx), 512 bf16 per fragment.
__device__ inline v16bf load_b_frag(const __bf16* frag, int lane) {
  v16bf b;
#pragma unroll
  for (int i = 0; i < 16; ++i) b[i] = frag[lane * 16 + i];
  return b;
}

// ------------------------------------------------- B-fragment pack (weights)
// W is [K x N] row-major f32. One block per (ntile,kfrag), 512 threads.
// Layout: lane L -> N = nt*16 + L%16 ; K = kf*32 + (v/4)*16 + (L/16)*8 + (v%4)*2 + p
__global__ void prep_bfrag(const float* __restrict__ W, int K, int N, int kfrags,
                           __bf16* __restrict__ dst) {
  int nt = blockIdx.x / kfrags, kf = blockIdx.x % kfrags;
  int t   = threadIdx.x;
  int L   = t >> 4;
  int idx = t & 15;
  int v = idx >> 1, p = idx & 1;
  int n = nt * 16 + (L & 15);
  int k = kf * 32 + (v >> 2) * 16 + (L >> 4) * 8 + (v & 3) * 2 + p;
  float val = (k < K && n < N) ? W[(size_t)k * N + n] : 0.0f;
  dst[(size_t)blockIdx.x * 512 + L * 16 + idx] = (__bf16)val;
}

// ------------------------------------------- item + user embeddings + codes
__global__ void item_user_kernel(const int* __restrict__ uid, const int* __restrict__ u1,
                                 const int* __restrict__ u2,  const int* __restrict__ u3,
                                 const int* __restrict__ u4,
                                 const int* __restrict__ ig,  const int* __restrict__ is,
                                 const int* __restrict__ ic,
                                 const float* __restrict__ emb, const float* __restrict__ H,
                                 float* __restrict__ xitem, unsigned* __restrict__ codes,
                                 float* __restrict__ xcat) {
  int b = blockIdx.x, t = threadIdx.x;
  __shared__ float xi[DIT_];
  __shared__ unsigned codeAcc;
  if (t == 0) codeAcc = 0u;
  if (t < DIT_) {
    int f = t / E_, d = t % E_;
    int id = (f == 0 ? ig : f == 1 ? is : ic)[b];
    float v = emb[(size_t)id * E_ + d];
    xi[t] = v;
    xitem[b * DIT_ + t] = v;
    xcat[(size_t)b * DIN_ + t] = v;
  } else if (t < 128) {
    int u = t - DIT_;
    int f = u / E_, d = u % E_;
    int id = (f == 0 ? uid : f == 1 ? u1 : f == 2 ? u2 : f == 3 ? u3 : u4)[b];
    xcat[(size_t)b * DIN_ + DIT_ + u] = emb[(size_t)id * E_ + d];
  }
  __syncthreads();
  if (t < 16) {
    float s = 0.0f;
    for (int e = 0; e < DIT_; ++e) s += xi[e] * H[e * 16 + t];
    if (s > 0.0f) atomicOr(&codeAcc, 1u << t);
  }
  __syncthreads();
  if (t == 0) codes[b] = codeAcc;
}

// --------------------------- fused long gather + LSH WMMA projection + score
// One wave (32 threads) per (b, tile of 16 long positions). Rows padded to 64
// columns (cols 48..63 zero) so A-fragment loads are branch-free.
__global__ void lsh_score_kernel(const int* __restrict__ lg, const int* __restrict__ ls,
                                 const int* __restrict__ lc,
                                 const float* __restrict__ emb,
                                 const unsigned* __restrict__ codes,
                                 const __bf16* __restrict__ Hfrag,
                                 float* __restrict__ scores) {
  int b = blockIdx.x, tile = blockIdx.y, lane = threadIdx.x;
  __shared__ __align__(16) float rows[16 * 64];
  __shared__ int gid_row[16];
  int base = b * LL_ + tile * 16;

  // stage 16 rows x 4 fields (field 3 = zero padding); each field = 1 cacheline
  for (int seg = lane; seg < 64; seg += 32) {
    int r = seg >> 2, f = seg & 3;
    float4* d4 = (float4*)(&rows[r * 64 + f * E_]);
    if (f == 3) {
      float4 z = make_float4(0.f, 0.f, 0.f, 0.f);
      d4[0] = z; d4[1] = z; d4[2] = z; d4[3] = z;
    } else {
      int id = (f == 0 ? lg : f == 1 ? ls : lc)[base + r];
      if (f == 0) gid_row[r] = id;
      const float4* s4 = (const float4*)(emb + (size_t)id * E_);
      d4[0] = s4[0]; d4[1] = s4[1]; d4[2] = s4[2]; d4[3] = s4[3];
    }
  }
  __syncthreads();

  // proj[16 pos x 16 hash] = rows[16x48] @ H[48x16]  (two K=32 bf16 WMMA steps)
  v8f acc = {};
  {
    v16bf a0 = load_a_frag(rows, 64, 0, lane);
    v16bf b0 = load_b_frag(Hfrag + 0 * 512, lane);
    acc = __builtin_amdgcn_wmma_f32_16x16x32_bf16(false, a0, false, b0, (short)0, acc, false, false);
    v16bf a1 = load_a_frag(rows, 64, 32, lane);
    v16bf b1 = load_b_frag(Hfrag + 1 * 512, lane);
    acc = __builtin_amdgcn_wmma_f32_16x16x32_bf16(false, a1, false, b1, (short)0, acc, false, false);
  }

  // sign-compare with item code bits, popcount matches -> score per position
  unsigned code = codes[b];
  int n = lane & 15;
  int cb = (int)((code >> n) & 1u);
#pragma unroll
  for (int v = 0; v < 8; ++v) {
    int pred  = (acc[v] > 0.0f) ? 1 : 0;
    unsigned m = wave_ballot32(pred == cb);
    if (lane == 0) {
      float s0 = (gid_row[v]     == 0) ? -__builtin_inff() : (float)__popc(m & 0xFFFFu);
      float s1 = (gid_row[v + 8] == 0) ? -__builtin_inff() : (float)__popc(m >> 16);
      scores[(size_t)b * LL_ + tile * 16 + v]     = s0;
      scores[(size_t)b * LL_ + tile * 16 + v + 8] = s1;
    }
  }
}

// ----------------------------------------------- bucketed top-48 of integers
__global__ void topk_kernel(const float* __restrict__ scores, int* __restrict__ topIdx) {
  __shared__ int hist[17];
  __shared__ int cntGT, cntEQ;
  __shared__ int thrS, needS;
  int b = blockIdx.x, t = threadIdx.x;
  if (t < 17) hist[t] = 0;
  if (t == 0) { cntGT = 0; cntEQ = 0; }
  __syncthreads();
  const float* sc = scores + (size_t)b * LL_;
  for (int i = t; i < LL_; i += 256) {
    float s = sc[i];
    if (s >= 0.0f) atomicAdd(&hist[(int)s], 1);
  }
  __syncthreads();
  if (t == 0) {
    int total = 0, thr = -1, need = 0;
    for (int s = 16; s >= 0; --s) {
      if (total + hist[s] >= TOPK_) { thr = s; need = TOPK_ - total; break; }
      total += hist[s];
    }
    thrS = thr; needS = need;
  }
  __syncthreads();
  int thr = thrS;
  for (int i = t; i < LL_; i += 256) {
    float s = sc[i];
    if (s < 0.0f) continue;
    if (thr < 0 || (int)s > thr) {
      int pos = atomicAdd(&cntGT, 1);
      topIdx[b * TOPK_ + pos] = i;
    }
  }
  __syncthreads();
  int baseGT = cntGT;
  if (thr >= 0) {
    for (int i = t; i < LL_; i += 256) {
      float s = sc[i];
      if (s >= 0.0f && (int)s == thr) {
        int pos = atomicAdd(&cntEQ, 1);
        if (pos < needS) topIdx[b * TOPK_ + baseGT + pos] = i;
      }
    }
  }
  __syncthreads();
  int filled = (thr >= 0) ? TOPK_ : baseGT;
  if (t < TOPK_ && t >= filled) topIdx[b * TOPK_ + t] = -1;
}

// ------------------------------------------------- single-query MHA (folded)
__device__ void attn_core(const float* Xs, int K, int klen, const float* xi,
                          const float* Wq, const float* bq,
                          const float* Wk, const float* bk,
                          const float* Wv, const float* bv,
                          const float* Wo, const float* bo,
                          float* qh, float* wtil, float* cb8, float* sc,
                          float* aX, float* od, float* dst) {
  int t = threadIdx.x; // 128 threads
  if (t < 48) {                 // qh[h*6+d] = xi . Wq[:,h,d] + bq
    float s = bq[t];
    for (int e = 0; e < 48; ++e) s += xi[e] * Wq[e * 48 + t];
    qh[t] = s;
  }
  __syncthreads();
  for (int o = t; o < 384; o += 128) {   // wtil[h][e] = sum_d Wk[e,h,d]*qh[h,d]
    int h = o / 48, e = o % 48;
    float s = 0.0f;
    for (int d = 0; d < KD_; ++d) s += Wk[e * 48 + h * KD_ + d] * qh[h * KD_ + d];
    wtil[o] = s;
  }
  if (t < 8) {                  // bias term of k-projection folded into score
    float s = 0.0f;
    for (int d = 0; d < KD_; ++d) s += bk[t * KD_ + d] * qh[t * KD_ + d];
    cb8[t] = s;
  }
  __syncthreads();
  const float rs = 0.40824829046386302f; // 1/sqrt(6)
  for (int k = t; k < K; k += 128) {
    for (int h = 0; h < NH_; ++h) {
      float s = cb8[h];
      for (int e = 0; e < 48; ++e) s += Xs[k * 48 + e] * wtil[h * 48 + e];
      s *= rs;
      if (k >= klen) s -= 1e9f;
      sc[h * K + k] = s;
    }
  }
  __syncthreads();
  if (t < 8) {                  // softmax per head
    float mx = -__builtin_inff();
    for (int k = 0; k < K; ++k) mx = fmaxf(mx, sc[t * K + k]);
    float sum = 0.0f;
    for (int k = 0; k < K; ++k) { float e = expf(sc[t * K + k] - mx); sc[t * K + k] = e; sum += e; }
    float inv = 1.0f / sum;
    for (int k = 0; k < K; ++k) sc[t * K + k] *= inv;
  }
  __syncthreads();
  for (int o = t; o < 384; o += 128) {   // aX[h][e] = sum_k a[h,k]*Xs[k,e]
    int h = o / 48, e = o % 48;
    float s = 0.0f;
    for (int k = 0; k < K; ++k) s += sc[h * K + k] * Xs[k * 48 + e];
    aX[o] = s;
  }
  __syncthreads();
  if (t < 48) {                 // o[h,d] = aX[h] . Wv[:,h,d] + bv  (sum a == 1)
    int h = t / KD_;
    float s = bv[t];
    for (int e = 0; e < 48; ++e) s += aX[h * 48 + e] * Wv[e * 48 + t];
    od[t] = s;
  }
  __syncthreads();
  if (t < 48) {                 // out[e] = sum_hd o[hd]*Wo[hd,e] + bo
    float s = bo[t];
    for (int hd = 0; hd < 48; ++hd) s += od[hd] * Wo[hd * 48 + t];
    dst[t] = s;
  }
  __syncthreads();
}

__global__ void attn_kernel(const int* __restrict__ sg, const int* __restrict__ ss,
                            const int* __restrict__ scid,
                            const int* __restrict__ lg, const int* __restrict__ ls,
                            const int* __restrict__ lc,
                            const float* __restrict__ emb, const float* __restrict__ xitem,
                            const int* __restrict__ topIdx,
                            const float* sWq, const float* sbq, const float* sWk, const float* sbk,
                            const float* sWv, const float* sbv, const float* sWo, const float* sbo,
                            const float* lWq, const float* lbq, const float* lWk, const float* lbk,
                            const float* lWv, const float* lbv, const float* lWo, const float* lbo,
                            float* __restrict__ xcat) {
  int b = blockIdx.x, t = threadIdx.x;
  __shared__ __align__(16) float Xs[LS_ * DIT_];
  __shared__ float xi[48], qh[48], wtil[384], cb8[8], aX[384], od[48];
  __shared__ float scb[NH_ * LS_];
  __shared__ int lenS;

  if (t == 0) lenS = 0;
  if (t < 48) xi[t] = xitem[b * DIT_ + t];
  __syncthreads();

  // ---- short-term: gather 128 x 48 into LDS, count valid length
  for (int seg = t; seg < LS_ * 3; seg += 128) {
    int r = seg / 3, f = seg % 3;
    int id = (f == 0 ? sg : f == 1 ? ss : scid)[b * LS_ + r];
    if (f == 0 && id != 0) atomicAdd(&lenS, 1);
    const float4* s4 = (const float4*)(emb + (size_t)id * E_);
    float4* d4 = (float4*)(&Xs[r * DIT_ + f * E_]);
    d4[0] = s4[0]; d4[1] = s4[1]; d4[2] = s4[2]; d4[3] = s4[3];
  }
  __syncthreads();
  attn_core(Xs, LS_, lenS, xi, sWq, sbq, sWk, sbk, sWv, sbv, sWo, sbo,
            qh, wtil, cb8, scb, aX, od, xcat + (size_t)b * DIN_ + 128);

  // ---- long-term: gather top-48 rows, valid entries packed first
  if (t == 0) lenS = 0;
  __syncthreads();
  for (int seg = t; seg < TOPK_ * 3; seg += 128) {
    int r = seg / 3, f = seg % 3;
    int li = topIdx[b * TOPK_ + r];
    float4* d4 = (float4*)(&Xs[r * DIT_ + f * E_]);
    if (li < 0) {
      float4 z = make_float4(0.f, 0.f, 0.f, 0.f);
      d4[0] = z; d4[1] = z; d4[2] = z; d4[3] = z;
    } else {
      int id = (f == 0 ? lg : f == 1 ? ls : lc)[b * LL_ + li];
      if (f == 0) atomicAdd(&lenS, 1);
      const float4* s4 = (const float4*)(emb + (size_t)id * E_);
      d4[0] = s4[0]; d4[1] = s4[1]; d4[2] = s4[2]; d4[3] = s4[3];
    }
  }
  __syncthreads();
  attn_core(Xs, TOPK_, lenS, xi, lWq, lbq, lWk, lbk, lWv, lbv, lWo, lbo,
            qh, wtil, cb8, scb, aX, od, xcat + (size_t)b * DIN_ + 176);
}

// -------------------------------------------------------- bf16 WMMA GEMM
// C[M x N] = A[M x Ka] @ W (prepacked B frags) + bias.  grid.x = M/16, 256 thr.
// KFRAGS is compile-time -> fully unrolled chain of WMMAs with clustered
// B-fragment loads. A tile staged to LDS with zero-padded stride KFRAGS*32.
template <int KFRAGS>
__global__ void gemm_wmma(const float* __restrict__ A, int Ka, int lda,
                          const __bf16* __restrict__ Bf, int ntiles,
                          const float* __restrict__ bias,
                          float* __restrict__ C, int N, int ldc) {
  constexpr int KPAD = KFRAGS * 32;
  __shared__ __align__(16) float rows[16 * KPAD];
  int mt = blockIdx.x, t = threadIdx.x;
  for (int i = t; i < 16 * KPAD; i += 256) {
    int r = i / KPAD, k = i % KPAD;
    rows[r * KPAD + k] = (k < Ka) ? A[(size_t)(mt * 16 + r) * lda + k] : 0.0f;
  }
  __syncthreads();
  int wave = t >> 5, lane = t & 31;
  for (int nt = wave; nt < ntiles; nt += 8) {
    const __bf16* bfrag = Bf + (size_t)nt * KFRAGS * 512;
    v8f acc = {};
#pragma unroll
    for (int kf = 0; kf < KFRAGS; ++kf) {
      v16bf a = load_a_frag(rows, KPAD, kf * 32, lane);
      v16bf b = load_b_frag(bfrag + kf * 512, lane);
      acc = __builtin_amdgcn_wmma_f32_16x16x32_bf16(false, a, false, b, (short)0, acc, false, false);
    }
    int n = nt * 16 + (lane & 15);
    if (n < N) {
      int mbase = mt * 16 + 8 * (lane >> 4);
      float bi = bias[n];
#pragma unroll
      for (int v = 0; v < 8; ++v) C[(size_t)(mbase + v) * ldc + n] = acc[v] + bi;
    }
  }
}

// -------------------------------------------------------------- LN + ReLU
__global__ void ln_relu_kernel(const float* __restrict__ X, const float* __restrict__ g,
                               const float* __restrict__ be, float* __restrict__ Y, int D) {
  __shared__ float s1[256], s2[256];
  int b = blockIdx.x, t = threadIdx.x;
  float a = 0.f, aa = 0.f;
  for (int i = t; i < D; i += 256) { float x = X[(size_t)b * D + i]; a += x; aa += x * x; }
  s1[t] = a; s2[t] = aa;
  __syncthreads();
  for (int st = 128; st > 0; st >>= 1) {
    if (t < st) { s1[t] += s1[t + st]; s2[t] += s2[t + st]; }
    __syncthreads();
  }
  float m = s1[0] / (float)D;
  float var = s2[0] / (float)D - m * m;
  float inv = rsqrtf(var + 1e-3f);
  for (int i = t; i < D; i += 256) {
    float x = X[(size_t)b * D + i];
    float y = (x - m) * inv * g[i] + be[i];
    Y[(size_t)b * D + i] = fmaxf(y, 0.0f);
  }
}

// ---------------------------------------------- LN + ReLU + 80->1 + sigmoid
__global__ void final_kernel(const float* __restrict__ h2, const float* __restrict__ g2,
                             const float* __restrict__ be2, const float* __restrict__ W3,
                             const float* __restrict__ b3, float* __restrict__ out) {
  __shared__ float s1[128], s2[128], red[128];
  int b = blockIdx.x, t = threadIdx.x;
  float x = (t < 80) ? h2[(size_t)b * 80 + t] : 0.0f;
  s1[t] = (t < 80) ? x : 0.f;
  s2[t] = (t < 80) ? x * x : 0.f;
  __syncthreads();
  for (int st = 64; st > 0; st >>= 1) {
    if (t < st) { s1[t] += s1[t + st]; s2[t] += s2[t + st]; }
    __syncthreads();
  }
  float m = s1[0] / 80.0f;
  float var = s2[0] / 80.0f - m * m;
  float inv = rsqrtf(var + 1e-3f);
  float contrib = 0.0f;
  if (t < 80) {
    float y = fmaxf((x - m) * inv * g2[t] + be2[t], 0.0f);
    contrib = y * W3[t];
  }
  red[t] = contrib;
  __syncthreads();
  for (int st = 64; st > 0; st >>= 1) {
    if (t < st) red[t] += red[t + st];
    __syncthreads();
  }
  if (t == 0) out[b] = 1.0f / (1.0f + expf(-(red[0] + b3[0])));
}

// ---------------------------------------------------------------- launcher
extern "C" void kernel_launch(void* const* d_in, const int* in_sizes, int n_in,
                              void* d_out, int out_size, void* d_ws, size_t ws_size,
                              hipStream_t stream) {
  (void)in_sizes; (void)n_in; (void)out_size; (void)ws_size;
  const int* uid = (const int*)d_in[0];
  const int* ut1 = (const int*)d_in[1];
  const int* ut2 = (const int*)d_in[2];
  const int* ut3 = (const int*)d_in[3];
  const int* ut4 = (const int*)d_in[4];
  const int* ig  = (const int*)d_in[5];
  const int* is  = (const int*)d_in[6];
  const int* ic  = (const int*)d_in[7];
  const int* sg  = (const int*)d_in[8];
  const int* ss  = (const int*)d_in[9];
  const int* scd = (const int*)d_in[10];
  const int* lg  = (const int*)d_in[11];
  const int* lsh = (const int*)d_in[12];
  const int* lcd = (const int*)d_in[13];
  const float* emb = (const float*)d_in[14];
  const float* H   = (const float*)d_in[15];
  const float* sWq = (const float*)d_in[16]; const float* sbq = (const float*)d_in[17];
  const float* sWk = (const float*)d_in[18]; const float* sbk = (const float*)d_in[19];
  const float* sWv = (const float*)d_in[20]; const float* sbv = (const float*)d_in[21];
  const float* sWo = (const float*)d_in[22]; const float* sbo = (const float*)d_in[23];
  const float* lWq = (const float*)d_in[24]; const float* lbq = (const float*)d_in[25];
  const float* lWk = (const float*)d_in[26]; const float* lbk = (const float*)d_in[27];
  const float* lWv = (const float*)d_in[28]; const float* lbv = (const float*)d_in[29];
  const float* lWo = (const float*)d_in[30]; const float* lbo = (const float*)d_in[31];
  const float* W1  = (const float*)d_in[32]; const float* b1  = (const float*)d_in[33];
  const float* g1  = (const float*)d_in[34]; const float* be1 = (const float*)d_in[35];
  const float* W2  = (const float*)d_in[36]; const float* b2  = (const float*)d_in[37];
  const float* g2v = (const float*)d_in[38]; const float* be2 = (const float*)d_in[39];
  const float* W3  = (const float*)d_in[40]; const float* b3  = (const float*)d_in[41];
  float* out = (float*)d_out;

  char* w = (char*)d_ws;
  size_t off = 0;
  auto alloc = [&](size_t bytes) -> void* {
    void* p = w + off;
    off = (off + bytes + 255) & ~(size_t)255;
    return p;
  };
  __bf16*   Hf     = (__bf16*)alloc(2 * 512 * sizeof(__bf16));          // 1 nt x 2 kf
  __bf16*   W1f    = (__bf16*)alloc(13 * 7 * 512 * sizeof(__bf16));     // 13 nt x 7 kf
  __bf16*   W2f    = (__bf16*)alloc(5 * 7 * 512 * sizeof(__bf16));      // 5 nt x 7 kf
  float*    xitem  = (float*)alloc((size_t)B_ * DIT_ * sizeof(float));
  unsigned* codes  = (unsigned*)alloc((size_t)B_ * sizeof(unsigned));
  float*    xcat   = (float*)alloc((size_t)B_ * DIN_ * sizeof(float));
  float*    scores = (float*)alloc((size_t)B_ * LL_ * sizeof(float));
  int*      topIdx = (int*)alloc((size_t)B_ * TOPK_ * sizeof(int));
  float*    h1pre  = (float*)alloc((size_t)B_ * 200 * sizeof(float));
  float*    h1a    = (float*)alloc((size_t)B_ * 200 * sizeof(float));
  float*    h2pre  = (float*)alloc((size_t)B_ * 80 * sizeof(float));

  // pack B fragments (constants, re-derived every call for determinism)
  prep_bfrag<<<1 * 2, 512, 0, stream>>>(H, 48, 16, 2, Hf);
  prep_bfrag<<<13 * 7, 512, 0, stream>>>(W1, 224, 200, 7, W1f);
  prep_bfrag<<<5 * 7, 512, 0, stream>>>(W2, 200, 80, 7, W2f);

  item_user_kernel<<<B_, 128, 0, stream>>>(uid, ut1, ut2, ut3, ut4, ig, is, ic,
                                           emb, H, xitem, codes, xcat);

  dim3 gLsh(B_, LL_ / 16);
  lsh_score_kernel<<<gLsh, 32, 0, stream>>>(lg, lsh, lcd, emb, codes, Hf, scores);

  topk_kernel<<<B_, 256, 0, stream>>>(scores, topIdx);

  attn_kernel<<<B_, 128, 0, stream>>>(sg, ss, scd, lg, lsh, lcd, emb, xitem, topIdx,
                                      sWq, sbq, sWk, sbk, sWv, sbv, sWo, sbo,
                                      lWq, lbq, lWk, lbk, lWv, lbv, lWo, lbo, xcat);

  gemm_wmma<7><<<B_ / 16, 256, 0, stream>>>(xcat, DIN_, DIN_, W1f, 13, b1, h1pre, 200, 200);
  ln_relu_kernel<<<B_, 256, 0, stream>>>(h1pre, g1, be1, h1a, 200);
  gemm_wmma<7><<<B_ / 16, 256, 0, stream>>>(h1a, 200, 200, W2f, 5, b2, h2pre, 80, 80);
  final_kernel<<<B_, 128, 0, stream>>>(h2pre, g2v, be2, W3, b3, out);
}
